// ContrastiveLoss_66795331388029
// MI455X (gfx1250) — matrix operand order repaired and checked by reading, not compiled, not run on hardware
//
#include <hip/hip_runtime.h>
#include <hip/hip_bf16.h>
#include <math.h>

// ---------------------------------------------------------------------------
// ContrastiveLoss (NT-Xent with diagonal labels) fused kernel for gfx1250.
// N = 8192 rows, D = 256. bf16 WMMA 16x16x32 with fp32 accumulate.
// ---------------------------------------------------------------------------

#define NROWS   8192
#define BHALF   4096
#define DIM     256
#define KF      8            // 256 / 32 K-fragments per tile
#define MT      2            // row tiles (16 rows each) per wave -> 32 rows/wave
#define CSPLIT  8            // column splits (partial exp-sums are additive)
#define CT      ((NROWS / CSPLIT) / 16)   // 64 column tiles per wave

typedef __attribute__((ext_vector_type(16))) __bf16 v16bf;
typedef __attribute__((ext_vector_type(8)))  __bf16 v8bf;
typedef __attribute__((ext_vector_type(8)))  float  v8f;
typedef __attribute__((ext_vector_type(4)))  float  v4f;

// ---------------------------------------------------------------------------
// Kernel 1: row-normalize projections -> bf16 Q; zero the per-row sum array.
// 256 threads/block, 32 rows/block (8 threads per row, each owning a
// contiguous 32-element chunk -> b128 loads, b128 bf16 stores).
// ---------------------------------------------------------------------------
__global__ void k_norm_bf16(const float* __restrict__ p1,
                            const float* __restrict__ p2,
                            __bf16* __restrict__ q,
                            float* __restrict__ s_global) {
  const int row   = blockIdx.x * 32 + (threadIdx.x >> 3);
  const int lane8 = threadIdx.x & 7;
  const float* src = (row < BHALF) ? (p1 + (size_t)row * DIM)
                                   : (p2 + (size_t)(row - BHALF) * DIM);
  // contiguous 32-element chunk per thread (128 B) -> vector loads
  const v4f* s4 = reinterpret_cast<const v4f*>(src + lane8 * 32);
  v4f vals[8];
  float ss = 0.f;
#pragma unroll
  for (int i = 0; i < 8; ++i) {
    v4f v = s4[i];
    vals[i] = v;
    ss += v.x * v.x + v.y * v.y + v.z * v.z + v.w * v.w;
  }
  // reduce sum of squares across the aligned 8-lane group
#pragma unroll
  for (int off = 1; off < 8; off <<= 1) ss += __shfl_xor(ss, off, 32);

  const float rn = rsqrtf(ss);      // norms ~16 here; never near zero/EPS
  __bf16* dst = q + (size_t)row * DIM + lane8 * 32;
#pragma unroll
  for (int i = 0; i < 8; ++i) {
    v4f v = vals[i];
    dst[i * 4 + 0] = (__bf16)(v.x * rn);
    dst[i * 4 + 1] = (__bf16)(v.y * rn);
    dst[i * 4 + 2] = (__bf16)(v.z * rn);
    dst[i * 4 + 3] = (__bf16)(v.w * rn);
  }

  if (threadIdx.x < 32) s_global[blockIdx.x * 32 + threadIdx.x] = 0.f;
}

// ---------------------------------------------------------------------------
// Kernel 2: fused Q*Q^T + exp-sum (fixed max = 2.0 since logits <= 2).
// Each wave: 2 row tiles (A held in VGPRs) x 1024-column slice.
// blockDim = 128 (4 waves), gridDim = 512 -> 2048 waves total.
// __launch_bounds__(128, 1): allow a full VGPR allocation so the 16
// persistent A fragments (128 VGPRs) are NOT spilled to scratch.
// ---------------------------------------------------------------------------
__global__ void __launch_bounds__(128, 1)
k_sim_lse(const __bf16* __restrict__ q, float* __restrict__ s_global) {
  const int lane    = threadIdx.x & 31;
  const int w       = blockIdx.x * 4 + (threadIdx.x >> 5);
  const int rowGrp  = w / CSPLIT;          // 0..255
  const int cs      = w % CSPLIT;          // 0..7
  const int rowBase = rowGrp * (MT * 16);
  const int laneLo  = lane & 15;
  const int laneHi  = lane >> 4;

  // ---- A fragments: 16x32 bf16 layout.
  // lanes 0-15: row M=lane, elems 0..7 = K ka..ka+7, elems 8..15 = K ka+16..+23
  // lanes 16-31: same rows, K offset +8 (ka = k*32 + laneHi*8)
  v16bf a[MT][KF];
#pragma unroll
  for (int m = 0; m < MT; ++m) {
    const __bf16* arow = q + (size_t)(rowBase + m * 16 + laneLo) * DIM;
#pragma unroll
    for (int k = 0; k < KF; ++k) {
      const int ka = k * 32 + laneHi * 8;
      v8bf lo = *reinterpret_cast<const v8bf*>(arow + ka);        // 16B aligned
      v8bf hi = *reinterpret_cast<const v8bf*>(arow + ka + 16);
      a[m][k] = __builtin_shufflevector(lo, hi, 0, 1, 2, 3, 4, 5, 6, 7,
                                        8, 9, 10, 11, 12, 13, 14, 15);
    }
  }

  float s[MT][8];
#pragma unroll
  for (int m = 0; m < MT; ++m)
#pragma unroll
    for (int v = 0; v < 8; ++v) s[m][v] = 0.f;

  const int jt0 = cs * CT;
  for (int jt = 0; jt < CT; ++jt) {
    const int j = jt0 + jt;
    // ---- B fragments: 32x16 bf16. lane holds column N=laneLo,
    // same K layout as A: elems 0..7 = K kb..kb+7, 8..15 = K kb+16..+23 (kb = k*32 + laneHi*8).
    const __bf16* brow = q + (size_t)(j * 16 + laneLo) * DIM;
    v16bf b[KF];
#pragma unroll
    for (int k = 0; k < KF; ++k) {
      v8bf blo = *reinterpret_cast<const v8bf*>(brow + k * 32 + laneHi * 8);
      v8bf bhi = *reinterpret_cast<const v8bf*>(brow + k * 32 + laneHi * 8 + 16);
      b[k] = __builtin_shufflevector(blo, bhi, 0, 1, 2, 3, 4, 5, 6, 7,
                                     8, 9, 10, 11, 12, 13, 14, 15);
    }

    const int col = j * 16 + laneLo;
#pragma unroll
    for (int m = 0; m < MT; ++m) {
      v8f acc = {};
#pragma unroll
      for (int k = 0; k < KF; ++k)
        acc = __builtin_amdgcn_wmma_f32_16x16x32_bf16(
            false, a[m][k], false, b[k], (short)0, acc, false, false);

      const int rb = rowBase + m * 16 + laneHi * 8;   // C/D: VGPR v -> row rb+v
      const int dv = col - rb;                        // diagonal at v == dv
#pragma unroll
      for (int v = 0; v < 8; ++v) {
        float x = 2.0f * acc[v] - 2.0f;               // logit - fixed_max(2.0)
        if (v == dv) x = -__builtin_inff();           // diagonal mask -> exp=0
        s[m][v] += __expf(x);
      }
    }
  }

  // reduce partial sums across the 16 lanes of each half-wave
#pragma unroll
  for (int m = 0; m < MT; ++m)
#pragma unroll
    for (int v = 0; v < 8; ++v)
#pragma unroll
      for (int off = 1; off < 16; off <<= 1)
        s[m][v] += __shfl_xor(s[m][v], off, 32);

  if (laneLo == 0) {
#pragma unroll
    for (int m = 0; m < MT; ++m) {
      const int rb = rowBase + m * 16 + laneHi * 8;
#pragma unroll
      for (int v = 0; v < 8; ++v) atomicAdd(&s_global[rb + v], s[m][v]);
    }
  }
}

// ---------------------------------------------------------------------------
// Kernel 3: loss = mean_i( -( logits[i,i] - lse_i ) ),  logits[i,i] = -2e9.
// Single block of 256 threads.
// ---------------------------------------------------------------------------
__global__ void k_loss(const float* __restrict__ s_global,
                       float* __restrict__ out) {
  __shared__ float red[256];
  float acc = 0.f;
  for (int r = threadIdx.x; r < NROWS; r += 256) {
    float lse = 2.0f + logf(s_global[r]);     // fixed max = 2.0
    float logp_ii = -2.0e9f - lse;            // masked diagonal logit - lse
    acc += -logp_ii;
  }
  red[threadIdx.x] = acc;
  __syncthreads();
  for (int st = 128; st > 0; st >>= 1) {
    if (threadIdx.x < st) red[threadIdx.x] += red[threadIdx.x + st];
    __syncthreads();
  }
  if (threadIdx.x == 0) out[0] = red[0] / (float)NROWS;
}

// ---------------------------------------------------------------------------
extern "C" void kernel_launch(void* const* d_in, const int* in_sizes, int n_in,
                              void* d_out, int out_size, void* d_ws, size_t ws_size,
                              hipStream_t stream) {
  (void)in_sizes; (void)n_in; (void)out_size; (void)ws_size;
  // inputs: [0]=embedding1 (unused), [1]=embedding2 (unused),
  //         [2]=projection1 [4096,256], [3]=projection2 [4096,256]
  const float* p1 = (const float*)d_in[2];
  const float* p2 = (const float*)d_in[3];
  float* out = (float*)d_out;

  // workspace layout: Q bf16 (8192*256*2 = 4 MB) | s_global (8192 f32)
  __bf16* q        = (__bf16*)d_ws;
  float*  s_global = (float*)((char*)d_ws + (size_t)NROWS * DIM * sizeof(__bf16));

  k_norm_bf16<<<NROWS / 32, 256, 0, stream>>>(p1, p2, q, s_global);

  const int waves  = (NROWS / (MT * 16)) * CSPLIT;  // 2048
  k_sim_lse<<<waves / 4, 128, 0, stream>>>(q, s_global);

  k_loss<<<1, 256, 0, stream>>>(s_global, out);
}
